// ScatterAndAvg3D_31353261261001
// MI455X (gfx1250) — matrix operand order repaired and checked by reading, not compiled, not run on hardware
//
#include <hip/hip_runtime.h>
#include <hip/hip_bf16.h>
#include <math.h>

// ---------------------------------------------------------------------------
// Problem constants (from the reference)
// ---------------------------------------------------------------------------
#define RES   64
#define CH    8
#define P3    27
#define PD    216           // true projection dim
#define PDP   224           // padded to multiple of 16 for WMMA N-tiles
#define DIN   64
#define BB    2
#define SS    8192
#define MROWS (BB * SS)     // 16384
#define N1    (PD * 8)      // 1728
#define N2    (PD * 4)      // 864
#define NVOX  (BB * RES * RES * RES)   // 524288 voxels

typedef __bf16 v16bf __attribute__((ext_vector_type(16)));
typedef __bf16 v8bf  __attribute__((ext_vector_type(8)));
typedef float  v8f   __attribute__((ext_vector_type(8)));

// ---------------------------------------------------------------------------
// CDNA5 async global->LDS copy (GLOBAL_LOAD_ASYNC_TO_LDS_B128, ASYNCcnt)
// ---------------------------------------------------------------------------
#if defined(__has_builtin)
#  if __has_builtin(__builtin_amdgcn_global_load_async_to_lds_b128)
#    define HAVE_ASYNC_LDS 1
#  endif
#endif

#ifdef HAVE_ASYNC_LDS
// builtin expects: (int4 AS1*, int4 AS3*, imm offset, imm cpol)
// (clang prints AS1 as '__device__' in HIP diagnostics)
typedef int gv4i __attribute__((vector_size(16)));
typedef __attribute__((address_space(1))) gv4i glb_v4i_t;
typedef __attribute__((address_space(3))) gv4i lds_v4i_t;
static __device__ __forceinline__ void async_copy_b128(const void* g, void* l) {
    // generic LDS pointer: low 32 bits are the LDS byte offset
    __builtin_amdgcn_global_load_async_to_lds_b128(
        (glb_v4i_t*)(uintptr_t)g,
        (lds_v4i_t*)(unsigned)(uintptr_t)l,
        0, 0);
}
#  if __has_builtin(__builtin_amdgcn_s_wait_asynccnt)
#    define WAIT_ASYNC() __builtin_amdgcn_s_wait_asynccnt(0)
#  else
#    define WAIT_ASYNC() asm volatile("s_wait_asynccnt 0x0" ::: "memory")
#  endif
#else
#  define WAIT_ASYNC() ((void)0)
#endif

// round-to-nearest-even f32 -> bf16
static __device__ __forceinline__ __bf16 f2bf(float f) {
    unsigned u = __float_as_uint(f);
    u += 0x7FFFu + ((u >> 16) & 1u);
    unsigned short h = (unsigned short)(u >> 16);
    __bf16 r;
    __builtin_memcpy(&r, &h, 2);
    return r;
}

static __device__ __forceinline__ float gelu_exact(float x) {
    return 0.5f * x * (1.0f + erff(x * 0.70710678118654752f));
}

// ---------------------------------------------------------------------------
// f32 -> bf16 conversion kernels
// ---------------------------------------------------------------------------
__global__ __launch_bounds__(256) void cvt_bf16_kernel(const float* __restrict__ in,
                                                       __bf16* __restrict__ out, int n) {
    int i = blockIdx.x * 256 + threadIdx.x;
    if (i < n) out[i] = f2bf(in[i]);
}

// Wf [rows x 216] f32  ->  [rows x 224] bf16, zero-padded columns
__global__ __launch_bounds__(256) void cvt_pad_bf16_kernel(const float* __restrict__ in,
                                                           __bf16* __restrict__ out,
                                                           int rows, int ncin, int ncout) {
    int i = blockIdx.x * 256 + threadIdx.x;
    int total = rows * ncout;
    if (i >= total) return;
    int r = i / ncout, c = i - r * ncout;
    out[i] = (c < ncin) ? f2bf(in[r * ncin + c]) : (__bf16)0.0f;
}

// ---------------------------------------------------------------------------
// bf16 WMMA GEMM:  C[M,N] = act(A[M,K] @ B[K,N] + bias)
//   block tile: 256(M) x 32(N) x 32(K), 256 threads = 8 waves
//   each wave: 32x32 tile = 2x2 v_wmma_f32_16x16x32_bf16
//   double-buffered LDS; A staged with GLOBAL_LOAD_ASYNC_TO_LDS_B128
// ---------------------------------------------------------------------------
template <bool GELU, bool OUT_BF16>
__global__ __launch_bounds__(256) void gemm_bf16_wmma(
    const __bf16* __restrict__ A, const __bf16* __restrict__ Bm,
    const float* __restrict__ bias, void* __restrict__ outp,
    int M, int N, int K, int Nreal)
{
    constexpr int BM = 256, BN = 32, BK = 32;
    constexpr int LDA = BK + 8;   // 80B row stride: 16B aligned, breaks bank-conflict strides
    constexpr int LDB = BK + 8;
    constexpr int ASZ = BM * LDA; // 10240 elems
    constexpr int BSZ = BN * LDB; // 1280 elems

    __shared__ __align__(16) __bf16 sA[2][ASZ];
    __shared__ __align__(16) __bf16 sB[2][BSZ];

    const int tid  = threadIdx.x;
    const int lane = tid & 31;
    const int wave = tid >> 5;                 // 0..7 -> 32-row M sub-tile
    const int half = lane >> 4;                // 0/1 (K-half select)
    const int lr   = lane & 15;

    const int m0 = blockIdx.y * BM;
    const int n0 = blockIdx.x * BN;

    // ---- per-thread staging coordinates (hoisted) ----
    const int ar0 = tid >> 2;                  // A rows: ar0 + {0,64,128,192}
    const int akc = (tid & 3) * 8;             // A k-chunk within tile
    const bool bload = tid < 128;
    const int bk  = tid >> 2;                  // B row within k-tile (tid<128)
    const int bn  = (tid & 3) * 8;             // B col chunk

    const int nk = K / BK;

    v8f acc[2][2] = {};
    v8bf breg;
#ifndef HAVE_ASYNC_LDS
    v8bf areg[4];
#endif

    // ================= prologue: stage tile 0 into buffer 0 =================
    {
#ifdef HAVE_ASYNC_LDS
        #pragma unroll
        for (int i = 0; i < 4; ++i)
            async_copy_b128(A + (size_t)(m0 + ar0 + i * 64) * K + akc,
                            &sA[0][(ar0 + i * 64) * LDA + akc]);
#else
        #pragma unroll
        for (int i = 0; i < 4; ++i)
            areg[i] = *(const v8bf*)(A + (size_t)(m0 + ar0 + i * 64) * K + akc);
#endif
        if (bload)
            breg = *(const v8bf*)(Bm + (size_t)bk * N + n0 + bn);
#ifndef HAVE_ASYNC_LDS
        #pragma unroll
        for (int i = 0; i < 4; ++i)
            *(v8bf*)(&sA[0][(ar0 + i * 64) * LDA + akc]) = areg[i];
#endif
        if (bload) {
            #pragma unroll
            for (int e = 0; e < 8; ++e) sB[0][(bn + e) * LDB + bk] = breg[e];
        }
        WAIT_ASYNC();
        __syncthreads();
    }

    // ================= main loop =================
    for (int ki = 0; ki < nk; ++ki) {
        const int cur = ki & 1, nxt = cur ^ 1;
        const bool has_next = (ki + 1) < nk;
        const int kb_next = (ki + 1) * BK;

        // ---- issue next-tile loads (overlap with WMMA below) ----
        if (has_next) {
#ifdef HAVE_ASYNC_LDS
            #pragma unroll
            for (int i = 0; i < 4; ++i)
                async_copy_b128(A + (size_t)(m0 + ar0 + i * 64) * K + kb_next + akc,
                                &sA[nxt][(ar0 + i * 64) * LDA + akc]);
#else
            #pragma unroll
            for (int i = 0; i < 4; ++i)
                areg[i] = *(const v8bf*)(A + (size_t)(m0 + ar0 + i * 64) * K + kb_next + akc);
#endif
            if (bload)
                breg = *(const v8bf*)(Bm + (size_t)(kb_next + bk) * N + n0 + bn);
        }

        // ---- compute on current buffers: build frags per ISA 7.12.2 ----
        {
            const __bf16* bufA = sA[cur];
            const __bf16* bufB = sB[cur];
            v16bf af[2], bf_[2];
            #pragma unroll
            for (int mt = 0; mt < 2; ++mt) {
                const __bf16* p = &bufA[(wave * 32 + mt * 16 + lr) * LDA + half * 8];
                v8bf lo = *(const v8bf*)p;          // K = kbase..kbase+7
                v8bf hi = *(const v8bf*)(p + 16);   // K = kbase+16..kbase+23
                af[mt] = __builtin_shufflevector(lo, hi,
                         0,1,2,3,4,5,6,7,8,9,10,11,12,13,14,15);
            }
            #pragma unroll
            for (int nt = 0; nt < 2; ++nt) {
                const __bf16* p = &bufB[(nt * 16 + lr) * LDB + half * 8];
                v8bf lo = *(const v8bf*)p;          // K = kbase..kbase+7
                v8bf hi = *(const v8bf*)(p + 16);   // K = kbase+16..kbase+23
                bf_[nt] = __builtin_shufflevector(lo, hi,
                          0,1,2,3,4,5,6,7,8,9,10,11,12,13,14,15);
            }
            #pragma unroll
            for (int mt = 0; mt < 2; ++mt)
                #pragma unroll
                for (int nt = 0; nt < 2; ++nt)
                    acc[mt][nt] = __builtin_amdgcn_wmma_f32_16x16x32_bf16(
                        false, af[mt], false, bf_[nt], (short)0, acc[mt][nt], false, false);
        }

        // ---- commit next-tile register-staged data to LDS ----
        if (has_next) {
#ifndef HAVE_ASYNC_LDS
            #pragma unroll
            for (int i = 0; i < 4; ++i)
                *(v8bf*)(&sA[nxt][(ar0 + i * 64) * LDA + akc]) = areg[i];
#endif
            if (bload) {
                #pragma unroll
                for (int e = 0; e < 8; ++e) sB[nxt][(bn + e) * LDB + bk] = breg[e];
            }
        }
        WAIT_ASYNC();
        __syncthreads();
    }

    // ---- epilogue: bias (+GELU), store bf16 or f32 ----
    float* outF  = (float*)outp;
    __bf16* outB = (__bf16*)outp;
    #pragma unroll
    for (int mt = 0; mt < 2; ++mt) {
        int rbase = m0 + wave * 32 + mt * 16 + half * 8;
        #pragma unroll
        for (int nt = 0; nt < 2; ++nt) {
            int col = n0 + nt * 16 + lr;
            float bcol = (col < Nreal) ? bias[col] : 0.0f;
            #pragma unroll
            for (int v = 0; v < 8; ++v) {
                float val = acc[mt][nt][v] + bcol;
                if (GELU) val = gelu_exact(val);
                size_t idx = (size_t)(rbase + v) * N + col;
                if (OUT_BF16) outB[idx] = f2bf(val);
                else          outF[idx] = val;
            }
        }
    }
}

// ---------------------------------------------------------------------------
// Scatter: feats[p, j] -> vox[b, clip(pos+off(j/8)), ch=j/27] ; counts for j<27
// (exactly reproduces the reference's repeat pairing and meshgrid 'xy' order)
// ---------------------------------------------------------------------------
__global__ __launch_bounds__(256) void scatter_kernel(
    const float* __restrict__ feats,      // [MROWS x PDP] padded
    const float* __restrict__ positions,  // [B,S,3]
    float* __restrict__ vox,              // [NVOX * CH]
    float* __restrict__ cnt)              // [NVOX]
{
    int tid = blockIdx.x * 256 + threadIdx.x;
    const int total = MROWS * PD;
    if (tid >= total) return;

    int p = tid / PD;
    int j = tid - p * PD;
    int b = p / SS;

    float fx = positions[p * 3 + 0];
    float fy = positions[p * 3 + 1];
    float fz = positions[p * 3 + 2];
    int px = (int)(fx * (float)RES);
    int py = (int)(fy * (float)RES);
    int pz = (int)(fz * (float)RES);

    int o  = j >> 3;        // offset index = j // CH
    int ch = j / P3;        // channel      = j // 27
    // meshgrid 'xy': offset o = i*9 + jj*3 + k -> (comp0, comp1, comp2) = (jj-1, i-1, k-1)
    int ox = (o / 3) % 3 - 1;
    int oy =  o / 9      - 1;
    int oz =  o % 3      - 1;

    int x0 = min(max(px + ox, 0), RES - 1);
    int y0 = min(max(py + oy, 0), RES - 1);
    int z0 = min(max(pz + oz, 0), RES - 1);

    int vlin = ((b * RES + x0) * RES + y0) * RES + z0;
    atomicAdd(&vox[(size_t)vlin * CH + ch], feats[(size_t)p * PDP + j]);
    if (j < P3) atomicAdd(&cnt[vlin], 1.0f);
}

// ---------------------------------------------------------------------------
// Pointwise dense: out[v, :216] = [vox[v,0:8], cnt[v]] @ Wp + bp
// 224 threads/block, 16 voxels/block; Wp column kept in registers.
// ---------------------------------------------------------------------------
__global__ __launch_bounds__(224) void voxel_dense_kernel(
    const float* __restrict__ vox, const float* __restrict__ cnt,
    const float* __restrict__ Wp,  const float* __restrict__ bp,
    float* __restrict__ out)
{
    constexpr int VPB = 16;
    __shared__ float sv[VPB][9];

    int n = threadIdx.x;                    // 0..223
    int vbase = blockIdx.x * VPB;

    if (n < VPB * 9) {
        int v = n / 9, c = n - v * 9;
        int voxel = vbase + v;
        sv[v][c] = (c < CH) ? vox[(size_t)voxel * CH + c] : cnt[voxel];
    }

    float w[9];
    float bias = 0.0f;
    const bool act = (n < PD);
    if (act) {
        #pragma unroll
        for (int c = 0; c < 9; ++c) w[c] = Wp[c * PD + n];   // Wp: [9 x 216] row-major
        bias = bp[n];
    }
    __syncthreads();

    if (act) {
        #pragma unroll 4
        for (int v = 0; v < VPB; ++v) {
            float s = bias;
            #pragma unroll
            for (int c = 0; c < 9; ++c) s = fmaf(sv[v][c], w[c], s);
            out[(size_t)(vbase + v) * PD + n] = s;
        }
    }
}

// ---------------------------------------------------------------------------
// Launch
// ---------------------------------------------------------------------------
static inline size_t align256(size_t x) { return (x + 255) & ~size_t(255); }

extern "C" void kernel_launch(void* const* d_in, const int* in_sizes, int n_in,
                              void* d_out, int out_size, void* d_ws, size_t ws_size,
                              hipStream_t stream) {
    const float* positions = (const float*)d_in[0];  // [B,S,3]
    const float* x         = (const float*)d_in[1];  // [B,S,64]
    const float* W1        = (const float*)d_in[2];  // [64,1728]
    const float* b1        = (const float*)d_in[3];
    const float* W2        = (const float*)d_in[4];  // [1728,864]
    const float* b2        = (const float*)d_in[5];
    const float* Wf        = (const float*)d_in[6];  // [864,216]
    const float* bf        = (const float*)d_in[7];
    const float* Wp        = (const float*)d_in[8];  // [9,216]
    const float* bp        = (const float*)d_in[9];
    float* out             = (float*)d_out;

    // ---- workspace carve-up ----
    char* ws = (char*)d_ws;
    size_t off = 0;
    __bf16* Xbf   = (__bf16*)(ws + off); off = align256(off + (size_t)MROWS * DIN * 2);
    __bf16* W1bf  = (__bf16*)(ws + off); off = align256(off + (size_t)DIN * N1 * 2);
    __bf16* H1    = (__bf16*)(ws + off); off = align256(off + (size_t)MROWS * N1 * 2);
    __bf16* W2bf  = (__bf16*)(ws + off); off = align256(off + (size_t)N1 * N2 * 2);
    __bf16* H2    = (__bf16*)(ws + off); off = align256(off + (size_t)MROWS * N2 * 2);
    __bf16* Wfbf  = (__bf16*)(ws + off); off = align256(off + (size_t)N2 * PDP * 2);
    float*  feats = (float*) (ws + off); off = align256(off + (size_t)MROWS * PDP * 4);
    float*  vox   = (float*) (ws + off); off = align256(off + (size_t)NVOX * CH * 4);
    float*  cnt   = (float*) (ws + off); off = align256(off + (size_t)NVOX * 4);
    (void)ws_size; (void)in_sizes; (void)n_in; (void)out_size;

    // ---- zero scatter targets (graph-capture-safe stream memset) ----
    (void)hipMemsetAsync(vox, 0, (size_t)NVOX * CH * 4, stream);
    (void)hipMemsetAsync(cnt, 0, (size_t)NVOX * 4, stream);

    // ---- bf16 conversions ----
    {
        int n = MROWS * DIN;
        cvt_bf16_kernel<<<(n + 255) / 256, 256, 0, stream>>>(x, Xbf, n);
        n = DIN * N1;
        cvt_bf16_kernel<<<(n + 255) / 256, 256, 0, stream>>>(W1, W1bf, n);
        n = N1 * N2;
        cvt_bf16_kernel<<<(n + 255) / 256, 256, 0, stream>>>(W2, W2bf, n);
        n = N2 * PDP;
        cvt_pad_bf16_kernel<<<(n + 255) / 256, 256, 0, stream>>>(Wf, Wfbf, N2, PD, PDP);
    }

    // ---- MLP: 3 WMMA GEMMs ----
    gemm_bf16_wmma<true,  true ><<<dim3(N1 / 32,  MROWS / 256), 256, 0, stream>>>(
        Xbf, W1bf, b1, H1, MROWS, N1, DIN, N1);
    gemm_bf16_wmma<true,  true ><<<dim3(N2 / 32,  MROWS / 256), 256, 0, stream>>>(
        H1, W2bf, b2, H2, MROWS, N2, N1, N2);
    gemm_bf16_wmma<false, false><<<dim3(PDP / 32, MROWS / 256), 256, 0, stream>>>(
        H2, Wfbf, bf, feats, MROWS, PDP, N2, PD);

    // ---- scatter ----
    {
        int total = MROWS * PD;
        scatter_kernel<<<(total + 255) / 256, 256, 0, stream>>>(feats, positions, vox, cnt);
    }

    // ---- pointwise voxel dense ----
    voxel_dense_kernel<<<NVOX / 16, 224, 0, stream>>>(vox, cnt, Wp, bp, out);
}
